// MultiHeadedLocalAttention_73409581023559
// MI455X (gfx1250) — matrix operand order repaired and checked by reading, not compiled, not run on hardware
//
#include <hip/hip_runtime.h>
#include <hip/hip_bf16.h>

// ---------------------------------------------------------------------------
// MultiHeadedLocalAttention for MI455X (gfx1250, wave32, WMMA bf16)
//   B=2, M=2048, D=1024, H=16, DH=64, WINDOW=16
// Pipeline: cvt(bf16) -> 3x WMMA GEMM proj (async-LDS double-buffered) ->
//           banded attention (WMMA) -> WMMA GEMM output projection (f32+bias)
// ---------------------------------------------------------------------------

#define BATCH   2
#define MSEQ    2048
#define DMODEL  1024
#define NHEADS  16
#define DHEAD   64
#define WIN     16
#define BM      (BATCH * MSEQ)          // 4096 rows

#define BKT     32                      // K tile (one WMMA K-step)
#define LDSROW  40                      // padded LDS row stride in halves (80 B)

typedef __attribute__((ext_vector_type(16))) __bf16 v16bf;
typedef __attribute__((ext_vector_type(8)))  float  v8f;

union FragB16 {
  v16bf  v;
  __bf16 e[16];
  uint4  q[2];
};

// ---- CDNA5 async global->LDS copy (ASYNCcnt-tracked) ----------------------
// VDST operand = 32-bit LDS byte address; for an LDS-aperture flat address,
// addr[31:0] IS the LDS offset (ISA 10.2 aperture rules).
__device__ __forceinline__ void async_ld_b128(const __bf16* g, __bf16* l) {
  unsigned loff = (unsigned)(uintptr_t)(void*)l;
  asm volatile("global_load_async_to_lds_b128 %0, %1, off"
               :: "v"(loff), "v"(g) : "memory");
}
__device__ __forceinline__ void wait_async0() {
  asm volatile("s_wait_asynccnt 0x0" ::: "memory");
}

// ---------------------------------------------------------------------------
// float -> bf16 elementwise convert
// ---------------------------------------------------------------------------
__global__ void cvt_f32_bf16(const float* __restrict__ x, __bf16* __restrict__ y, int n) {
  int i = blockIdx.x * blockDim.x + threadIdx.x;
  if (i < n) y[i] = (__bf16)x[i];
}

// Wt[n*D + k] = (bf16) W[k*D + n]   (transpose so B-operand loads are contiguous)
__global__ void transpose_cvt_bf16(const float* __restrict__ W, __bf16* __restrict__ Wt) {
  int i = blockIdx.x * blockDim.x + threadIdx.x;   // i over D*D
  int nrow = i >> 10;
  int k    = i & (DMODEL - 1);
  Wt[i] = (__bf16)W[k * DMODEL + nrow];
}

// ---------------------------------------------------------------------------
// C[M,N] = A[M,K] @ Bt[N,K]^T + bias   via V_WMMA_F32_16X16X32_BF16
// Block: 256 thr = 8 waves, tile 128x128. Wave: 32x64 (2x4 accumulators).
// A/B tiles staged in LDS with async copies, double-buffered.
// outf != nullptr -> f32 output, else bf16 output.
// ---------------------------------------------------------------------------
__global__ __launch_bounds__(256)
void gemm_bf16_wmma(const __bf16* __restrict__ A, const __bf16* __restrict__ Bt,
                    const float* __restrict__ bias,
                    __bf16* __restrict__ outb, float* __restrict__ outf,
                    int Mrows, int N, int K) {
  __shared__ __align__(16) __bf16 As[2][128][LDSROW];
  __shared__ __align__(16) __bf16 Bs[2][128][LDSROW];

  const int tid  = threadIdx.x;
  const int lane = tid & 31;
  const int wid  = tid >> 5;
  const int nb   = N >> 7;                       // 128-wide col tiles
  const int bm   = (blockIdx.x / nb) << 7;       // block row base
  const int bn   = (blockIdx.x % nb) << 7;       // block col base
  const int wm   = (wid >> 1) << 5;              // wave row offset in tile
  const int wn   = (wid & 1) << 6;               // wave col offset in tile
  const int ln   = lane & 15;
  const int kbA  = (lane >> 4) << 4;             // A frag K-base: 0 / 16
  const int kbB  = (lane >> 4) << 4;             // B frag K-base: 0 / 16
  if (bm >= Mrows) return;

  v8f acc[2][4] = {};

  // ---- prologue: stage K-tile 0 into buffer 0 (512 16B chunks per matrix)
#pragma unroll
  for (int i = 0; i < 2; ++i) {
    int cid = tid + i * 256;
    int row = cid >> 2, kq = (cid & 3) << 3;     // kq in halves (0,8,16,24)
    async_ld_b128(A  + (size_t)(bm + row) * K + kq, &As[0][row][kq]);
    async_ld_b128(Bt + (size_t)(bn + row) * K + kq, &Bs[0][row][kq]);
  }

  int buf = 0;
  for (int k0 = 0; k0 < K; k0 += BKT) {
    wait_async0();          // our async writes for `buf` are in LDS
    __syncthreads();        // everyone's writes visible; prev reads of buf^1 done

    if (k0 + BKT < K) {     // stage next K-tile into the other buffer
      int k1 = k0 + BKT;
#pragma unroll
      for (int i = 0; i < 2; ++i) {
        int cid = tid + i * 256;
        int row = cid >> 2, kq = (cid & 3) << 3;
        async_ld_b128(A  + (size_t)(bm + row) * K + k1 + kq, &As[buf ^ 1][row][kq]);
        async_ld_b128(Bt + (size_t)(bn + row) * K + k1 + kq, &Bs[buf ^ 1][row][kq]);
      }
    }

    FragB16 bfr[4];
#pragma unroll
    for (int nt = 0; nt < 4; ++nt) {
      const __bf16* bp = &Bs[buf][wn + nt * 16 + ln][kbB];
      bfr[nt].q[0] = *(const uint4*)bp;
      bfr[nt].q[1] = *(const uint4*)(bp + 8);
    }
#pragma unroll
    for (int mt = 0; mt < 2; ++mt) {
      FragB16 a;
      const __bf16* ap = &As[buf][wm + mt * 16 + ln][kbA];
      a.q[0] = *(const uint4*)ap;
      a.q[1] = *(const uint4*)(ap + 8);
#pragma unroll
      for (int nt = 0; nt < 4; ++nt)
        acc[mt][nt] = __builtin_amdgcn_wmma_f32_16x16x32_bf16(
            false, a.v, false, bfr[nt].v, (short)0, acc[mt][nt], false, false);
    }
    buf ^= 1;
  }

  const int r0 = (lane >> 4) << 3;               // C/D row base: 0 / 8
#pragma unroll
  for (int mt = 0; mt < 2; ++mt) {
#pragma unroll
    for (int nt = 0; nt < 4; ++nt) {
      int col = bn + wn + nt * 16 + ln;
      float bv = bias[col];
#pragma unroll
      for (int r = 0; r < 8; ++r) {
        size_t idx = (size_t)(bm + wm + mt * 16 + r0 + r) * N + col;
        float val = acc[mt][nt][r] + bv;
        if (outf) outf[idx] = val;
        else      outb[idx] = (__bf16)val;
      }
    }
  }
}

// ---------------------------------------------------------------------------
// Banded local attention with adaptive top-16 threshold + softmax.
// One wave per (b, h, 16-query tile). Key window: 64 keys [i0-16, i0+48).
// q/k/v/ctx layouts: [B*M, D] bf16 where (m, h*64+dh).
// ---------------------------------------------------------------------------
__global__ __launch_bounds__(32)
void local_attn_wmma(const __bf16* __restrict__ qh, const __bf16* __restrict__ kh,
                     const __bf16* __restrict__ vh, __bf16* __restrict__ ctx) {
  __shared__ __align__(16) float  Sl[16][68];   // scores (padded row scans)
  __shared__ __align__(16) __bf16 Pl[16][72];   // probs (144B stride, b128-aligned)

  const int lane = threadIdx.x & 31;
  const int ln   = lane & 15;
  const int kbA  = (lane >> 4) << 4;
  const int kbB  = (lane >> 4) << 4;
  const int r0   = (lane >> 4) << 3;

  int t = blockIdx.x;
  const int qtiles = MSEQ / 16;
  const int qt = t % qtiles;  t /= qtiles;
  const int h  = t % NHEADS;
  const int b  = t / NHEADS;
  const int i0     = qt * 16;
  const int kstart = i0 - 16;

  const size_t hoff  = (size_t)h * DHEAD;
  const size_t bbase = (size_t)b * MSEQ;

  // ---- S[16,64] = Q_tile @ K_tile^T
  v8f s[4] = {};
#pragma unroll
  for (int d0 = 0; d0 < DHEAD; d0 += 32) {
    FragB16 a;
    const __bf16* qp = qh + (bbase + i0 + ln) * DMODEL + hoff + d0 + kbA;
    a.q[0] = *(const uint4*)(qp);
    a.q[1] = *(const uint4*)(qp + 8);
#pragma unroll
    for (int nt = 0; nt < 4; ++nt) {
      int j  = kstart + nt * 16 + ln;
      int jc = j < 0 ? 0 : (j >= MSEQ ? MSEQ - 1 : j);
      const __bf16* kp = kh + (bbase + jc) * DMODEL + hoff + d0 + kbB;
      FragB16 bb;
      bb.q[0] = *(const uint4*)(kp);
      bb.q[1] = *(const uint4*)(kp + 8);
      s[nt] = __builtin_amdgcn_wmma_f32_16x16x32_bf16(
          false, a.v, false, bb.v, (short)0, s[nt], false, false);
    }
  }

  // scale (1/sqrt(64)) + band / bounds mask -> LDS
#pragma unroll
  for (int nt = 0; nt < 4; ++nt) {
    int j = kstart + nt * 16 + ln;
#pragma unroll
    for (int r = 0; r < 8; ++r) {
      int qrow = r0 + r;
      int dj   = j - (i0 + qrow);
      bool ok  = (j >= 0) && (j < MSEQ) && (dj < WIN) && (dj > -WIN);
      Sl[qrow][nt * 16 + ln] = ok ? s[nt][r] * 0.125f : -1e9f;
    }
  }
  __syncthreads();

  // ---- per-row adaptive top-16 threshold + softmax (lane == row)
  if (lane < 16) {
    float m0 = -3e38f;
#pragma unroll
    for (int j = 0; j < 64; ++j) m0 = fmaxf(m0, Sl[lane][j]);
    float th = m0;
#pragma unroll 1
    for (int it = 1; it < WIN; ++it) {       // 15 strict-max steps -> 16th largest
      float nm = -3e38f;
      for (int j = 0; j < 64; ++j) {
        float v = Sl[lane][j];
        nm = (v < th) ? fmaxf(nm, v) : nm;
      }
      th = nm;
    }
    float sum = 0.f;
    for (int j = 0; j < 64; ++j) {
      float v = Sl[lane][j];
      float p = (v >= th) ? __expf(v - m0) : 0.f;
      Sl[lane][j] = p;
      sum += p;
    }
    float rinv = 1.f / fmaxf(sum, 1e-20f);
    for (int j = 0; j < 64; ++j)
      Pl[lane][j] = (__bf16)(Sl[lane][j] * rinv);
  }
  __syncthreads();

  // ---- ctx[16,64] = P[16,64] @ V[64,64]
  v8f c[4] = {};
#pragma unroll
  for (int k0 = 0; k0 < 64; k0 += 32) {
    FragB16 a;
    a.q[0] = *(const uint4*)(&Pl[ln][k0 + kbA]);
    a.q[1] = *(const uint4*)(&Pl[ln][k0 + kbA + 8]);
#pragma unroll
    for (int nt = 0; nt < 4; ++nt) {
      int n = nt * 16 + ln;
      FragB16 bb;
#pragma unroll
      for (int u = 0; u < 16; ++u) {
        int kk = kstart + k0 + kbB + u;
        int kc = kk < 0 ? 0 : (kk >= MSEQ ? MSEQ - 1 : kk);
        bb.e[u] = vh[(bbase + kc) * DMODEL + hoff + n];
      }
      c[nt] = __builtin_amdgcn_wmma_f32_16x16x32_bf16(
          false, a.v, false, bb.v, (short)0, c[nt], false, false);
    }
  }

#pragma unroll
  for (int nt = 0; nt < 4; ++nt) {
    int n = nt * 16 + ln;
#pragma unroll
    for (int r = 0; r < 8; ++r) {
      int qi = i0 + r0 + r;
      ctx[(bbase + qi) * DMODEL + hoff + n] = (__bf16)c[nt][r];
    }
  }
}

// ---------------------------------------------------------------------------
extern "C" void kernel_launch(void* const* d_in, const int* in_sizes, int n_in,
                              void* d_out, int out_size, void* d_ws, size_t ws_size,
                              hipStream_t stream) {
  const float* kf = (const float*)d_in[0];
  const float* vf = (const float*)d_in[1];
  const float* qf = (const float*)d_in[2];
  // d_in[3] = mask (all ones) -- handled implicitly by the band mask
  const float* Wk = (const float*)d_in[4];
  const float* bk = (const float*)d_in[5];
  const float* Wv = (const float*)d_in[6];
  const float* bv = (const float*)d_in[7];
  const float* Wq = (const float*)d_in[8];
  const float* bq = (const float*)d_in[9];
  const float* Wo = (const float*)d_in[10];
  const float* bo = (const float*)d_in[11];
  float* out = (float*)d_out;

  const size_t XE = (size_t)BM * DMODEL;        // 4M elems
  const size_t WE = (size_t)DMODEL * DMODEL;    // 1M elems

  char* ws = (char*)d_ws;
  __bf16* kx  = (__bf16*)ws; ws += XE * 2;
  __bf16* vx  = (__bf16*)ws; ws += XE * 2;
  __bf16* qx  = (__bf16*)ws; ws += XE * 2;
  __bf16* WkT = (__bf16*)ws; ws += WE * 2;
  __bf16* WvT = (__bf16*)ws; ws += WE * 2;
  __bf16* WqT = (__bf16*)ws; ws += WE * 2;
  __bf16* WoT = (__bf16*)ws; ws += WE * 2;
  __bf16* khp = (__bf16*)ws; ws += XE * 2;
  __bf16* vhp = (__bf16*)ws; ws += XE * 2;
  __bf16* qhp = (__bf16*)ws; ws += XE * 2;
  __bf16* ctx = (__bf16*)ws; ws += XE * 2;

  // 1) convert inputs to bf16
  {
    int n = (int)XE, thr = 256, blk = (n + thr - 1) / thr;
    cvt_f32_bf16<<<blk, thr, 0, stream>>>(kf, kx, n);
    cvt_f32_bf16<<<blk, thr, 0, stream>>>(vf, vx, n);
    cvt_f32_bf16<<<blk, thr, 0, stream>>>(qf, qx, n);
  }
  // 2) transpose + convert weights
  {
    int n = (int)WE, thr = 256, blk = (n + thr - 1) / thr;
    transpose_cvt_bf16<<<blk, thr, 0, stream>>>(Wk, WkT);
    transpose_cvt_bf16<<<blk, thr, 0, stream>>>(Wv, WvT);
    transpose_cvt_bf16<<<blk, thr, 0, stream>>>(Wq, WqT);
    transpose_cvt_bf16<<<blk, thr, 0, stream>>>(Wo, WoT);
  }
  // 3) projections: [4096,1024] @ [1024,1024] + b  (bf16 out)
  {
    int blk = (BM / 128) * (DMODEL / 128);      // 256 blocks of 128x128 tiles
    gemm_bf16_wmma<<<blk, 256, 0, stream>>>(kx, WkT, bk, khp, nullptr, BM, DMODEL, DMODEL);
    gemm_bf16_wmma<<<blk, 256, 0, stream>>>(vx, WvT, bv, vhp, nullptr, BM, DMODEL, DMODEL);
    gemm_bf16_wmma<<<blk, 256, 0, stream>>>(qx, WqT, bq, qhp, nullptr, BM, DMODEL, DMODEL);
  }
  // 4) banded attention: one wave per (b, h, 16-query tile)
  {
    int blk = BATCH * NHEADS * (MSEQ / 16);     // 4096
    local_attn_wmma<<<blk, 32, 0, stream>>>(qhp, khp, vhp, ctx);
  }
  // 5) output projection: ctx @ Wo + bo -> f32 d_out
  {
    int blk = (BM / 128) * (DMODEL / 128);
    gemm_bf16_wmma<<<blk, 256, 0, stream>>>(ctx, WoT, bo, nullptr, out, BM, DMODEL, DMODEL);
  }
  (void)in_sizes; (void)n_in; (void)out_size; (void)ws_size;
}
